// BNO_61272003445400
// MI455X (gfx1250) — matrix operand order repaired and checked by reading, not compiled, not run on hardware
//
#include <hip/hip_runtime.h>
#include <hip/hip_bf16.h>
#include <math.h>

// ---------------------------------------------------------------------------
// CDNA5 (gfx1250) implementation of the branched Fourier neural operator.
// wave32; all heavy contractions via v_wmma_f32_16x16x32_bf16.
// ---------------------------------------------------------------------------

typedef __attribute__((ext_vector_type(16))) __bf16 v16bf;
typedef __attribute__((ext_vector_type(8)))  float  v8f;

#define QPAD 80     // 65 basis funcs padded to 5 col-tiles of 16
#define ABK  160    // back-proj K-dim: 96 (padded bases) + 64 (channel mix)
#define CCH  64     // channels
#define KM   32     // Fourier modes

static __device__ __forceinline__ unsigned short f2bfbits(float f) {
  unsigned int u = __builtin_bit_cast(unsigned int, f);
  unsigned int r = u + 0x7FFFu + ((u >> 16) & 1u);   // round-to-nearest-even
  return (unsigned short)(r >> 16);
}
static __device__ __forceinline__ float bf2f(unsigned short s) {
  return __builtin_bit_cast(float, (unsigned int)s << 16);
}
static __device__ __forceinline__ __bf16 bits2bf(unsigned short s) {
  return __builtin_bit_cast(__bf16, s);
}
static __device__ __forceinline__ float gelu_exact(float x) {
  return 0.5f * x * (1.0f + erff(x * 0.70710678118654752f));
}

// --------------------------- tiny prep kernels -----------------------------

// m[k*2+d] = modes[k,d,0] * (0.5 + 1.5*sigmoid(lat[d,0]))
__global__ void k_modes(const float* __restrict__ modes,
                        const float* __restrict__ lat,
                        float* __restrict__ m) {
  int idx = threadIdx.x;
  if (idx < KM * 2) {
    int d = idx & 1;
    float s = 1.0f / (1.0f + expf(-lat[d]));
    m[idx] = modes[idx] * (0.5f + 1.5f * s);
  }
}

// cos/sin(2*pi * n . m_k) as bf16 bits, layout (b*N + x)*K + k
__global__ void k_bases(const float* __restrict__ nodes,
                        const float* __restrict__ m,
                        unsigned short* __restrict__ bc,
                        unsigned short* __restrict__ bs,
                        int N) {
  size_t g = (size_t)blockIdx.x * blockDim.x + threadIdx.x;
  if (g >= (size_t)4 * N) return;
  float n0 = nodes[g * 2 + 0];
  float n1 = nodes[g * 2 + 1];
  for (int k = 0; k < KM; ++k) {
    float ph = 6.28318530717958648f * (n0 * m[k * 2 + 0] + n1 * m[k * 2 + 1]);
    bc[g * KM + k] = f2bfbits(cosf(ph));
    bs[g * KM + k] = f2bfbits(sinf(ph));
  }
}

// lifting: out[b,c,x] = sum_i w[c,i]*in[b,x,i] + bias[c]  (IN = 2 or 3)
__global__ void k_fc0(const float* __restrict__ in, const float* __restrict__ w,
                      const float* __restrict__ bias, float* __restrict__ out,
                      int N, int IN) {
  size_t g = (size_t)blockIdx.x * blockDim.x + threadIdx.x;
  if (g >= (size_t)4 * N) return;
  int b = (int)(g / N), x = (int)(g % N);
  float vi[3] = {0.f, 0.f, 0.f};
  for (int i = 0; i < IN; ++i) vi[i] = in[g * IN + i];
  for (int c = 0; c < CCH; ++c) {
    float a = bias[c];
    for (int i = 0; i < IN; ++i) a += w[c * IN + i] * vi[i];
    out[((size_t)b * CCH + c) * N + x] = a;
  }
}

// ------------------- forward moments (WMMA, split-K) -----------------------
// partials[(b*nslabs+slab)][ch][q] = sum_{x in slab} X[b,ch,x] * psi_q(x)
__global__ void k_moments_partial(const float* __restrict__ X,
                                  const unsigned short* __restrict__ basC,
                                  const unsigned short* __restrict__ basS,
                                  const float* __restrict__ wgt,
                                  float* __restrict__ partials,
                                  int N, int slabLen, int nslabs) {
  __shared__ unsigned short Xl[CCH][33];
  __shared__ unsigned short Pl[QPAD][33];
  int b = blockIdx.y, slab = blockIdx.x;
  int tid = threadIdx.x, lane = tid & 31, wv = tid >> 5;
  int mlo = lane & 15, hi = (lane >> 4) & 1;
  int x0 = slab * slabLen;
  v8f acc[5];
  v8f zero = {0.f, 0.f, 0.f, 0.f, 0.f, 0.f, 0.f, 0.f};
  for (int i = 0; i < 5; ++i) acc[i] = zero;

  for (int xs = x0; xs < x0 + slabLen; xs += 32) {
    __syncthreads();
    for (int idx = tid; idx < CCH * 32; idx += 128) {
      int ch = idx >> 5, xx = idx & 31;
      Xl[ch][xx] = f2bfbits(X[((size_t)b * CCH + ch) * N + (xs + xx)]);
    }
    for (int idx = tid; idx < QPAD * 32; idx += 128) {
      int q = idx >> 5, xx = idx & 31;
      size_t x = (size_t)b * N + (xs + xx);
      float ww = wgt[x];
      float phi;
      if (q < KM)            phi = bf2f(basC[x * KM + q]);
      else if (q < 2 * KM)   phi = bf2f(basS[x * KM + (q - KM)]);
      else if (q == 2 * KM)  phi = 1.0f;
      else                   phi = 0.0f;
      Pl[q][xx] = f2bfbits(phi * ww);
    }
    __syncthreads();
    // A fragment: this wave's 16 channel rows x 32 x-cols
    v16bf a;
    for (int e = 0; e < 16; ++e) {
      int kk = (e & 7) + (hi ? 8 : 0) + ((e >= 8) ? 16 : 0);
      a[e] = bits2bf(Xl[wv * 16 + mlo][kk]);
    }
    for (int ct = 0; ct < 5; ++ct) {
      v16bf bm;
      int q = ct * 16 + mlo;
      for (int e = 0; e < 16; ++e) {
        int kk = (e & 7) + (hi ? 8 : 0) + ((e >= 8) ? 16 : 0);
        bm[e] = bits2bf(Pl[q][kk]);
      }
      acc[ct] = __builtin_amdgcn_wmma_f32_16x16x32_bf16(
          false, a, false, bm, (short)0, acc[ct], false, false);
    }
  }
  size_t base = (size_t)(b * nslabs + slab) * CCH * QPAD;
  for (int ct = 0; ct < 5; ++ct) {
    int q = ct * 16 + mlo;
    for (int v = 0; v < 8; ++v) {
      int ch = wv * 16 + v + hi * 8;
      partials[base + (size_t)ch * QPAD + q] = acc[ct][v];
    }
  }
}

__global__ void k_mreduce(const float* __restrict__ partials,
                          float* __restrict__ M, int nslabs) {
  int idx = blockIdx.x * blockDim.x + threadIdx.x;
  if (idx >= 4 * CCH * QPAD) return;
  int b = idx / (CCH * QPAD);
  int rem = idx % (CCH * QPAD);
  float s = 0.f;
  for (int sl = 0; sl < nslabs; ++sl)
    s += partials[(size_t)(b * nslabs + sl) * CCH * QPAD + rem];
  M[idx] = s;
}

// -------------------------- mode mixing (small) ----------------------------
// Builds back-projection A matrix (64 x 160):
//   A[o, k]      = 2*sum_i (Mc*wc + Ms*ws)         (cos coeffs)
//   A[o, 32+k]   = 2*sum_i (Ms*wc - Mc*ws)         (sin coeffs)
//   A[o, 64]     = sum_i M0[i]*w0[i,o]             (DC)
//   A[o, 65..95] = 0                               (padding)
//   A[o, 96+i]   = wmix[o,i]                       (residual channel mix)
__global__ void k_mix(const float* __restrict__ M1, const float* __restrict__ wc1,
                      const float* __restrict__ ws1, const float* __restrict__ w01,
                      const float* __restrict__ M2, const float* __restrict__ wc2,
                      const float* __restrict__ ws2, const float* __restrict__ w02,
                      const float* __restrict__ wmix, float* __restrict__ A,
                      int has2) {
  int b = blockIdx.x, tid = threadIdx.x;
  const float* M1b = M1 + (size_t)b * CCH * QPAD;
  const float* M2b = M2 ? (M2 + (size_t)b * CCH * QPAD) : M1b;
  for (int idx = tid; idx < CCH * KM; idx += blockDim.x) {
    int o = idx >> 5, k = idx & 31;
    float gc = 0.f, gs = 0.f;
    for (int i = 0; i < CCH; ++i) {
      float mc = M1b[i * QPAD + k], ms = M1b[i * QPAD + KM + k];
      float wc = wc1[((size_t)i * CCH + o) * KM + k];
      float ws = ws1[((size_t)i * CCH + o) * KM + k];
      gc += mc * wc + ms * ws;
      gs += ms * wc - mc * ws;
    }
    if (has2) {
      for (int i = 0; i < CCH; ++i) {
        float mc = M2b[i * QPAD + k], ms = M2b[i * QPAD + KM + k];
        float wc = wc2[((size_t)i * CCH + o) * KM + k];
        float ws = ws2[((size_t)i * CCH + o) * KM + k];
        gc += mc * wc + ms * ws;
        gs += ms * wc - mc * ws;
      }
    }
    A[((size_t)b * CCH + o) * ABK + k]      = 2.0f * gc;
    A[((size_t)b * CCH + o) * ABK + KM + k] = 2.0f * gs;
  }
  for (int o = tid; o < CCH; o += blockDim.x) {
    float g0 = 0.f;
    for (int i = 0; i < CCH; ++i) g0 += M1b[i * QPAD + 2 * KM] * w01[i * CCH + o];
    if (has2)
      for (int i = 0; i < CCH; ++i) g0 += M2b[i * QPAD + 2 * KM] * w02[i * CCH + o];
    A[((size_t)b * CCH + o) * ABK + 2 * KM] = g0;
  }
  for (int idx = tid; idx < CCH * 31; idx += blockDim.x) {
    int o = idx / 31, q = 65 + idx % 31;
    A[((size_t)b * CCH + o) * ABK + q] = 0.0f;
  }
  for (int idx = tid; idx < CCH * CCH; idx += blockDim.x) {
    int o = idx >> 6, i = idx & 63;
    A[((size_t)b * CCH + o) * ABK + 96 + i] = wmix[o * CCH + i];
  }
}

// ---------------- fused back-projection + residual (WMMA) ------------------
// Out[b,o,x] = gelu?( A[b,o,:] . [cos;sin;1;0;Xcur[:,x]] + bias[o] )
__global__ void k_backproj(const float* __restrict__ A,
                           const unsigned short* __restrict__ basC,
                           const unsigned short* __restrict__ basS,
                           const float* __restrict__ Xcur,
                           const float* __restrict__ bias,
                           float* __restrict__ Out, int N, int applyGelu) {
  __shared__ unsigned short Bl[ABK][17];
  int b = blockIdx.y, chunk = blockIdx.x;           // 256 points per block
  int tid = threadIdx.x, lane = tid & 31, wv = tid >> 5;
  int mlo = lane & 15, hi = (lane >> 4) & 1;

  // Preload constant A fragments for the whole chunk (5 K-steps of 32)
  v16bf af[5];
  int row = wv * 16 + mlo;
  for (int ks = 0; ks < 5; ++ks)
    for (int e = 0; e < 16; ++e) {
      int kk = ks * 32 + (e & 7) + (hi ? 8 : 0) + ((e >= 8) ? 16 : 0);
      af[ks][e] = bits2bf(f2bfbits(A[((size_t)b * CCH + row) * ABK + kk]));
    }
  float bvals[8];
  for (int v = 0; v < 8; ++v) bvals[v] = bias[wv * 16 + v + hi * 8];

  v8f zero = {0.f, 0.f, 0.f, 0.f, 0.f, 0.f, 0.f, 0.f};
  for (int sub = 0; sub < 16; ++sub) {
    int x0 = chunk * 256 + sub * 16;
    __syncthreads();
    for (int idx = tid; idx < ABK * 16; idx += 128) {
      int r = idx >> 4, col = idx & 15;
      size_t x = (size_t)b * N + (x0 + col);
      float vv;
      if (r < KM)            vv = bf2f(basC[x * KM + r]);
      else if (r < 2 * KM)   vv = bf2f(basS[x * KM + (r - KM)]);
      else if (r == 2 * KM)  vv = 1.0f;
      else if (r < 96)       vv = 0.0f;
      else                   vv = Xcur[((size_t)b * CCH + (r - 96)) * N + (x0 + col)];
      Bl[r][col] = f2bfbits(vv);
    }
    __syncthreads();
    v8f acc = zero;
    for (int ks = 0; ks < 5; ++ks) {
      v16bf bm;
      for (int e = 0; e < 16; ++e)
        bm[e] = bits2bf(Bl[ks * 32 + (e & 7) + (hi ? 8 : 0) + ((e >= 8) ? 16 : 0)][mlo]);
      acc = __builtin_amdgcn_wmma_f32_16x16x32_bf16(
          false, af[ks], false, bm, (short)0, acc, false, false);
    }
    int x = x0 + mlo;
    for (int v = 0; v < 8; ++v) {
      int o = wv * 16 + v + hi * 8;
      float val = acc[v] + bvals[v];
      if (applyGelu) val = gelu_exact(val);
      Out[((size_t)b * CCH + o) * N + x] = val;
    }
  }
}

// -------------------------- fused fc1 + fc2 --------------------------------
__global__ void k_fc1fc2(const float* __restrict__ U, const float* __restrict__ w1,
                         const float* __restrict__ b1, const float* __restrict__ w2,
                         const float* __restrict__ b2, float* __restrict__ out,
                         int N) {
  __shared__ float W1[128 * CCH];
  __shared__ float B1[128];
  __shared__ float W2[128];
  int tid = threadIdx.x;
  for (int i = tid; i < 128 * CCH; i += 256) W1[i] = w1[i];
  for (int i = tid; i < 128; i += 256) { B1[i] = b1[i]; W2[i] = w2[i]; }
  __syncthreads();
  size_t g = (size_t)blockIdx.x * 256 + tid;
  if (g >= (size_t)4 * N) return;
  int b = (int)(g / N), x = (int)(g % N);
  float u[CCH];
  for (int c = 0; c < CCH; ++c) u[c] = U[((size_t)b * CCH + c) * N + x];
  float acc = 0.f;
  for (int f = 0; f < 128; ++f) {
    float h = B1[f];
    const float* wr = &W1[f * CCH];
    for (int c = 0; c < CCH; ++c) h += wr[c] * u[c];
    acc += W2[f] * gelu_exact(h);
  }
  out[g] = acc + b2[0];
}

// ---------------------------------------------------------------------------

extern "C" void kernel_launch(void* const* d_in, const int* in_sizes, int n_in,
                              void* d_out, int out_size, void* d_ws, size_t ws_size,
                              hipStream_t stream) {
  const int B = 4, NU = 65536, NV = 16384, SLAB = 4096;
  const float* u_in   = (const float*)d_in[0];
  const float* v_in   = (const float*)d_in[1];
  // d_in[2] = node_mask_u (unused by reference)
  const float* nodesU = (const float*)d_in[3];
  const float* nodesV = (const float*)d_in[4];
  const float* wgtU   = (const float*)d_in[5];
  const float* wgtV   = (const float*)d_in[6];
  const float* modes  = (const float*)d_in[7];
  const float* lat    = (const float*)d_in[8];
  const float* fc0uw  = (const float*)d_in[9];
  const float* fc0ub  = (const float*)d_in[10];
  const float* fc0vw  = (const float*)d_in[11];
  const float* fc0vb  = (const float*)d_in[12];
  const float* extwc  = (const float*)d_in[13];
  const float* extws  = (const float*)d_in[14];
  const float* extw0  = (const float*)d_in[15];
  const float* uwc    = (const float*)d_in[16];
  const float* uws    = (const float*)d_in[17];
  const float* uw0    = (const float*)d_in[18];
  const float* vwc    = (const float*)d_in[19];
  const float* vws    = (const float*)d_in[20];
  const float* vw0    = (const float*)d_in[21];
  const float* wsuW   = (const float*)d_in[22];
  const float* wsuB   = (const float*)d_in[23];
  const float* wsvW   = (const float*)d_in[24];
  const float* wsvB   = (const float*)d_in[25];
  const float* fc1w   = (const float*)d_in[26];
  const float* fc1b   = (const float*)d_in[27];
  const float* fc2w   = (const float*)d_in[28];
  const float* fc2b   = (const float*)d_in[29];

  char* ws = (char*)d_ws;
  size_t off = 0;
  auto arena = [&](size_t bytes) -> void* {
    void* p = ws + off;
    off = (off + bytes + 255) & ~(size_t)255;
    return p;
  };
  float* mscaled        = (float*)arena(64 * sizeof(float));
  unsigned short* basCU = (unsigned short*)arena((size_t)B * NU * KM * 2);
  unsigned short* basSU = (unsigned short*)arena((size_t)B * NU * KM * 2);
  unsigned short* basCV = (unsigned short*)arena((size_t)B * NV * KM * 2);
  unsigned short* basSV = (unsigned short*)arena((size_t)B * NV * KM * 2);
  float* U0 = (float*)arena((size_t)B * CCH * NU * 4);
  float* U1 = (float*)arena((size_t)B * CCH * NU * 4);
  float* V0 = (float*)arena((size_t)B * CCH * NV * 4);
  float* V1 = (float*)arena((size_t)B * CCH * NV * 4);
  float* Mu = (float*)arena((size_t)B * CCH * QPAD * 4);
  float* Mv = (float*)arena((size_t)B * CCH * QPAD * 4);
  float* partials = (float*)arena((size_t)16 * B * CCH * QPAD * 4);
  float* Au = (float*)arena((size_t)B * CCH * ABK * 4);
  float* Av = (float*)arena((size_t)B * CCH * ABK * 4);
  (void)ws_size; (void)in_sizes; (void)n_in; (void)out_size;

  k_modes<<<1, 64, 0, stream>>>(modes, lat, mscaled);
  k_bases<<<(B * NU + 255) / 256, 256, 0, stream>>>(nodesU, mscaled, basCU, basSU, NU);
  k_bases<<<(B * NV + 255) / 256, 256, 0, stream>>>(nodesV, mscaled, basCV, basSV, NV);
  k_fc0<<<(B * NU + 255) / 256, 256, 0, stream>>>(u_in, fc0uw, fc0ub, U0, NU, 2);
  k_fc0<<<(B * NV + 255) / 256, 256, 0, stream>>>(v_in, fc0vw, fc0vb, V0, NV, 3);

  float *Uc = U0, *Un = U1, *Vc = V0, *Vn = V1;
  const int nsU = NU / SLAB, nsV = NV / SLAB;
  const size_t sK = (size_t)CCH * CCH * KM;   // layer stride, wc/ws
  const size_t s0 = (size_t)CCH * CCH;        // layer stride, w0 / wmix
  const int redBlocks = (4 * CCH * QPAD + 255) / 256;

  for (int l = 0; l < 3; ++l) {
    // moments of v and u (split-K partials + deterministic reduction)
    k_moments_partial<<<dim3(nsV, B), 128, 0, stream>>>(Vc, basCV, basSV, wgtV,
                                                        partials, NV, SLAB, nsV);
    k_mreduce<<<redBlocks, 256, 0, stream>>>(partials, Mv, nsV);
    k_moments_partial<<<dim3(nsU, B), 128, 0, stream>>>(Uc, basCU, basSU, wgtU,
                                                        partials, NU, SLAB, nsU);
    k_mreduce<<<redBlocks, 256, 0, stream>>>(partials, Mu, nsU);

    // u-update: u1 (ext conv of v) + u2 (self conv of u) + u3 (channel mix)
    k_mix<<<B, 256, 0, stream>>>(Mv, extwc + l * sK, extws + l * sK, extw0 + l * s0,
                                 Mu, uwc + l * sK, uws + l * sK, uw0 + l * s0,
                                 wsuW + l * s0, Au, 1);
    k_backproj<<<dim3(NU / 256, B), 128, 0, stream>>>(Au, basCU, basSU, Uc,
                                                      wsuB + l * CCH, Un, NU,
                                                      (l < 2) ? 1 : 0);
    if (l < 2) {
      // v-update: v1 (self conv of v) + v2 (channel mix), always gelu'd
      k_mix<<<B, 256, 0, stream>>>(Mv, vwc + l * sK, vws + l * sK, vw0 + l * s0,
                                   nullptr, nullptr, nullptr, nullptr,
                                   wsvW + l * s0, Av, 0);
      k_backproj<<<dim3(NV / 256, B), 128, 0, stream>>>(Av, basCV, basSV, Vc,
                                                        wsvB + l * CCH, Vn, NV, 1);
      float* t = Vc; Vc = Vn; Vn = t;
    }
    float* t = Uc; Uc = Un; Un = t;
  }

  k_fc1fc2<<<(B * NU + 255) / 256, 256, 0, stream>>>(Uc, fc1w, fc1b, fc2w, fc2b,
                                                     (float*)d_out, NU);
}
